// AudioVisualModel_8804682956916
// MI455X (gfx1250) — matrix operand order repaired and checked by reading, not compiled, not run on hardware
//
#include <hip/hip_runtime.h>
#include <hip/hip_bf16.h>
#include <math.h>

// Problem sizes (fixed by the reference)
#define B_   24
#define NA_  50
#define T_   10
#define NV_  196
#define D_   512
#define M_   (B_ * NA_)        // 1200 audio rows
#define VR_  (B_ * T_ * NV_)   // 47040 visual rows
#define GRP_ (B_ * T_)         // 240 (y,t) groups
#define NT_  13                // 13 * 16 = 208 >= 196 padded columns per group

typedef __attribute__((ext_vector_type(16))) __bf16 v16bf;
typedef __attribute__((ext_vector_type(8)))  __bf16 v8bf;
typedef __attribute__((ext_vector_type(4)))  __bf16 v4bf;
typedef __attribute__((ext_vector_type(8)))  float  v8f;

union V16u { v16bf v; v8bf h[2]; };

// ---------------- Path A (big ws): byte offsets, all 256-aligned ------------
#define A_AHI   ((size_t)256)
#define A_ALO   ((size_t)(256 + 1228800))                      // + M_*D_*2
#define A_VHI   ((size_t)(256 + 2 * 1228800))
#define A_VLO   ((size_t)(256 + 2 * 1228800 + 48168960))       // + VR_*D_*2
#define A_MAXV  ((size_t)(256 + 2 * 1228800 + 2 * 48168960))
#define A_END   (A_MAXV + (size_t)288000 * 4)                  // 99,947,776 B

// ---------------- Path B (small ws): float offsets --------------------------
#define WS_NONNEG 0
#define WS_ARN    64
#define WS_VRN    (64 + 1216)
#define WS_MAXVIS (64 + 1216 + 47040 + 64)                     // + 288000 f32

__device__ __forceinline__ void split_bf16(float x, __bf16& h, __bf16& l) {
  h = (__bf16)x;
  l = (__bf16)(x - (float)h);
}

// ---------------------------------------------------------------------------
// Kernel 0: zero the nonneg accumulator (ws[0] on both paths)
// ---------------------------------------------------------------------------
__global__ void init_kernel(float* ws) {
  if (threadIdx.x == 0) ws[0] = 0.0f;
}

// ===========================================================================
// PATH A: normalize + bf16x2-split once, then a load/WMMA-only GEMM.
// ===========================================================================

// One wave per row: L2-normalize, split into bf16 hi/lo planes.
__global__ __launch_bounds__(32)
void normsplit_kernel(const float* __restrict__ audio,
                      const float* __restrict__ visual,
                      __bf16* __restrict__ a_hi, __bf16* __restrict__ a_lo,
                      __bf16* __restrict__ v_hi, __bf16* __restrict__ v_lo) {
  const int row  = blockIdx.x;
  const int lane = threadIdx.x;
  const bool isA = (row < M_);
  const int  r   = isA ? row : (row - M_);
  const float4* p4 = (const float4*)((isA ? audio : visual) + (size_t)r * D_);

  float4 x[4];
  float ss = 0.0f;
#pragma unroll
  for (int i = 0; i < 4; ++i) {
    x[i] = p4[lane + 32 * i];
    ss += x[i].x * x[i].x + x[i].y * x[i].y + x[i].z * x[i].z + x[i].w * x[i].w;
  }
#pragma unroll
  for (int m = 16; m >= 1; m >>= 1) ss += __shfl_xor(ss, m, 32);
  const float rn = 1.0f / fmaxf(sqrtf(ss), 1e-12f);

  __bf16* hp = (isA ? a_hi : v_hi) + (size_t)r * D_;
  __bf16* lp = (isA ? a_lo : v_lo) + (size_t)r * D_;
#pragma unroll
  for (int i = 0; i < 4; ++i) {
    float e[4] = {x[i].x * rn, x[i].y * rn, x[i].z * rn, x[i].w * rn};
    v4bf h4, l4;
#pragma unroll
    for (int c = 0; c < 4; ++c) { __bf16 h, l; split_bf16(e[c], h, l); h4[c] = h; l4[c] = l; }
    const int o = (lane + 32 * i) * 4;
    *(v4bf*)(hp + o) = h4;
    *(v4bf*)(lp + o) = l4;
  }
}

// Fused GEMM (loads land directly in WMMA layout) + max-over-Nv + clip^2.
// grid = (19, 240); block = 128 (4 waves); wave owns a 16-row M tile.
__global__ __launch_bounds__(128)
void simsmax_pre_kernel(const __bf16* __restrict__ a_hi,
                        const __bf16* __restrict__ a_lo,
                        const __bf16* __restrict__ v_hi,
                        const __bf16* __restrict__ v_lo,
                        const float* __restrict__ temp_p,
                        float* __restrict__ max_vis,
                        float* __restrict__ nonneg_acc) {
  const int lane = threadIdx.x & 31;
  const int wave = threadIdx.x >> 5;
  const int m0   = (blockIdx.x * 4 + wave) * 16;
  if (m0 >= M_) return;

  const int gy = blockIdx.y;
  const int yb = gy / T_;
  const int tt = gy % T_;
  const float invT = 1.0f / temp_p[0];

  const int half = lane >> 4;
  const int l16  = lane & 15;

  const __bf16* ahirow = a_hi + (size_t)(m0 + l16) * D_;
  const __bf16* alorow = a_lo + (size_t)(m0 + l16) * D_;
  const int kA = half * 16;                // A: K 0..15 / 16..31
  const int kB = half * 16;                // B: K 0..15 / 16..31, col = l16

  v8f acc[NT_];
  const v8f vzero = {};
#pragma unroll
  for (int j = 0; j < NT_; ++j) acc[j] = vzero;

  for (int kc = 0; kc < D_; kc += 32) {
    V16u ahi, alo;
    ahi.h[0] = *(const v8bf*)(ahirow + kc + kA);
    ahi.h[1] = *(const v8bf*)(ahirow + kc + kA + 8);
    alo.h[0] = *(const v8bf*)(alorow + kc + kA);
    alo.h[1] = *(const v8bf*)(alorow + kc + kA + 8);

#pragma unroll
    for (int j = 0; j < NT_; ++j) {
      const int vg = j * 16 + l16;
      v16bf bhi, blo;
      if (vg < NV_) {
        const size_t vo = ((size_t)gy * NV_ + vg) * D_ + kc + kB;  // 32B aligned
        bhi = *(const v16bf*)(v_hi + vo);
        blo = *(const v16bf*)(v_lo + vo);
      } else {
        V16u z; z.h[0] = (v8bf)(__bf16)0.0f; z.h[1] = (v8bf)(__bf16)0.0f;
        bhi = z.v; blo = z.v;
      }
      acc[j] = __builtin_amdgcn_wmma_f32_16x16x32_bf16(false, ahi.v, false, bhi,
                                                       (short)0, acc[j], false, false);
      acc[j] = __builtin_amdgcn_wmma_f32_16x16x32_bf16(false, ahi.v, false, blo,
                                                       (short)0, acc[j], false, false);
      acc[j] = __builtin_amdgcn_wmma_f32_16x16x32_bf16(false, alo.v, false, bhi,
                                                       (short)0, acc[j], false, false);
    }
  }

  // Epilogue (C layout: lane<16 -> rows e, lane>=16 -> rows e+8; col = l16)
  float nn = 0.0f;
  float rmax[8];
#pragma unroll
  for (int e = 0; e < 8; ++e) rmax[e] = -3.0e38f;
#pragma unroll
  for (int j = 0; j < NT_; ++j) {
    const bool valid = (j * 16 + l16) < NV_;
#pragma unroll
    for (int e = 0; e < 8; ++e) {
      float s = acc[j][e] * invT;
      float c = fminf(fmaxf(s, -20.0f), 0.0f);
      nn += c * c;                                 // padded cols: s==0 -> 0
      if (valid) rmax[e] = fmaxf(rmax[e], s);
    }
  }
#pragma unroll
  for (int e = 0; e < 8; ++e)
#pragma unroll
    for (int m = 8; m >= 1; m >>= 1)
      rmax[e] = fmaxf(rmax[e], __shfl_xor(rmax[e], m, 32));
#pragma unroll
  for (int m = 16; m >= 1; m >>= 1) nn += __shfl_xor(nn, m, 32);
  if (lane == 0) atomicAdd(nonneg_acc, nn);
  if (l16 == 0) {
#pragma unroll
    for (int e = 0; e < 8; ++e) {
      int m = m0 + half * 8 + e;
      int x = m / NA_, a = m % NA_;
      max_vis[(((size_t)x * B_ + yb) * NA_ + a) * T_ + tt] = rmax[e];
    }
  }
}

// ===========================================================================
// PATH B: small-workspace fallback (in-kernel conversion; unchanged logic).
// ===========================================================================
__global__ __launch_bounds__(32)
void norm_kernel(const float* __restrict__ audio,
                 const float* __restrict__ visual,
                 float* __restrict__ a_rn,
                 float* __restrict__ v_rn) {
  const int row  = blockIdx.x;
  const int lane = threadIdx.x;
  const float* p = (row < M_) ? (audio + (size_t)row * D_)
                              : (visual + (size_t)(row - M_) * D_);
  const float4* p4 = (const float4*)p;
  float ss = 0.0f;
#pragma unroll
  for (int i = 0; i < 4; ++i) {
    float4 v = p4[lane + 32 * i];
    ss += v.x * v.x + v.y * v.y + v.z * v.z + v.w * v.w;
  }
#pragma unroll
  for (int m = 16; m >= 1; m >>= 1) ss += __shfl_xor(ss, m, 32);
  if (lane == 0) {
    float rn = 1.0f / fmaxf(sqrtf(ss), 1e-12f);
    if (row < M_) a_rn[row] = rn;
    else          v_rn[row - M_] = rn;
  }
}

__global__ __launch_bounds__(128)
void simsmax_kernel(const float* __restrict__ audio,
                    const float* __restrict__ visual,
                    const float* __restrict__ a_rn,
                    const float* __restrict__ v_rn,
                    const float* __restrict__ temp_p,
                    float* __restrict__ max_vis,
                    float* __restrict__ nonneg_acc) {
  const int lane = threadIdx.x & 31;
  const int wave = threadIdx.x >> 5;
  const int m0   = (blockIdx.x * 4 + wave) * 16;
  if (m0 >= M_) return;
  const int gy = blockIdx.y;
  const int yb = gy / T_;
  const int tt = gy % T_;
  const float invT = 1.0f / temp_p[0];
  const int half = lane >> 4;
  const int l16  = lane & 15;
  const int   mrow = m0 + l16;
  const float arn  = a_rn[mrow];
  const float* arow = audio + (size_t)mrow * D_;
  const int kA = half * 16;
  const float* vbase = visual + (size_t)gy * (NV_ * D_);
  const int kB = half * 16;

  v8f acc[NT_];
  const v8f vzero = {};
#pragma unroll
  for (int j = 0; j < NT_; ++j) acc[j] = vzero;

  for (int kc = 0; kc < D_; kc += 32) {
    float4 a0 = *(const float4*)(arow + kc + kA + 0);
    float4 a1 = *(const float4*)(arow + kc + kA + 4);
    float4 a2 = *(const float4*)(arow + kc + kA + 8);
    float4 a3 = *(const float4*)(arow + kc + kA + 12);
    float af[16] = {a0.x, a0.y, a0.z, a0.w, a1.x, a1.y, a1.z, a1.w,
                    a2.x, a2.y, a2.z, a2.w, a3.x, a3.y, a3.z, a3.w};
    v16bf ahi, alo;
#pragma unroll
    for (int i = 0; i < 16; ++i) {
      __bf16 h, l; split_bf16(af[i] * arn, h, l); ahi[i] = h; alo[i] = l;
    }
#pragma unroll
    for (int j = 0; j < NT_; ++j) {
      const int vg = j * 16 + l16;
      v16bf bhi, blo;
      if (vg < NV_) {
        const float vrn = v_rn[gy * NV_ + vg];
        const float* vrow = vbase + (size_t)vg * D_ + kc + kB;
        float4 b0 = *(const float4*)(vrow + 0);
        float4 b1 = *(const float4*)(vrow + 4);
        float4 b2 = *(const float4*)(vrow + 8);
        float4 b3 = *(const float4*)(vrow + 12);
        float bf[16] = {b0.x, b0.y, b0.z, b0.w, b1.x, b1.y, b1.z, b1.w,
                        b2.x, b2.y, b2.z, b2.w, b3.x, b3.y, b3.z, b3.w};
#pragma unroll
        for (int i = 0; i < 16; ++i) {
          __bf16 h, l; split_bf16(bf[i] * vrn, h, l); bhi[i] = h; blo[i] = l;
        }
      } else {
#pragma unroll
        for (int i = 0; i < 16; ++i) { bhi[i] = (__bf16)0.0f; blo[i] = (__bf16)0.0f; }
      }
      acc[j] = __builtin_amdgcn_wmma_f32_16x16x32_bf16(false, ahi, false, bhi,
                                                       (short)0, acc[j], false, false);
      acc[j] = __builtin_amdgcn_wmma_f32_16x16x32_bf16(false, ahi, false, blo,
                                                       (short)0, acc[j], false, false);
      acc[j] = __builtin_amdgcn_wmma_f32_16x16x32_bf16(false, alo, false, bhi,
                                                       (short)0, acc[j], false, false);
    }
  }

  float nn = 0.0f;
  float rmax[8];
#pragma unroll
  for (int e = 0; e < 8; ++e) rmax[e] = -3.0e38f;
#pragma unroll
  for (int j = 0; j < NT_; ++j) {
    const bool valid = (j * 16 + l16) < NV_;
#pragma unroll
    for (int e = 0; e < 8; ++e) {
      float s = acc[j][e] * invT;
      float c = fminf(fmaxf(s, -20.0f), 0.0f);
      nn += c * c;
      if (valid) rmax[e] = fmaxf(rmax[e], s);
    }
  }
#pragma unroll
  for (int e = 0; e < 8; ++e)
#pragma unroll
    for (int m = 8; m >= 1; m >>= 1)
      rmax[e] = fmaxf(rmax[e], __shfl_xor(rmax[e], m, 32));
#pragma unroll
  for (int m = 16; m >= 1; m >>= 1) nn += __shfl_xor(nn, m, 32);
  if (lane == 0) atomicAdd(nonneg_acc, nn);
  if (l16 == 0) {
#pragma unroll
    for (int e = 0; e < 8; ++e) {
      int m = m0 + half * 8 + e;
      int x = m / NA_, a = m % NA_;
      max_vis[(((size_t)x * B_ + yb) * NA_ + a) * T_ + tt] = rmax[e];
    }
  }
}

// ---------------------------------------------------------------------------
// Finalize: temporal aggregation + InfoNCE + regularizers -> 5 scalars.
// ---------------------------------------------------------------------------
__global__ __launch_bounds__(256)
void finalize_kernel(const float* __restrict__ max_vis,
                     const float* __restrict__ nonneg_acc,
                     const float* __restrict__ temp_p,
                     const float* __restrict__ scale_p,
                     const float* __restrict__ thr_p,
                     float* __restrict__ out) {
  __shared__ float clip[B_ * B_];
  __shared__ float la[B_], lv[B_];
  __shared__ float cnt, srs;
  const int tid = threadIdx.x;

  for (int i = tid; i < B_ * B_; i += blockDim.x) clip[i] = 0.0f;
  if (tid == 0) { cnt = 0.0f; srs = 0.0f; }
  __syncthreads();

  const float th = 1.0f / (1.0f + expf(-thr_p[0]));
  const float sf = scale_p[0];

  float myCnt = 0.0f, mySrs = 0.0f;
  for (int i = tid; i < B_ * B_ * NA_; i += blockDim.x) {
    const int x = i / (B_ * NA_);
    const int r = i % (B_ * NA_);
    const int y = r / NA_;
    const float* mv = max_vis + (size_t)i * T_;
    float ws = 0.0f, wsum = 0.0f;
#pragma unroll
    for (int t = 0; t < T_; ++t) {
      float m   = mv[t];
      float raw = m - th;
      float sel = fmaxf(raw, 0.0f) * sf;
      ws += m * sel; wsum += sel;
      if (raw > 0.0f) myCnt += 1.0f;
      if (x == y)     mySrs += (tanhf(20.0f * raw) + 1.0f) * 0.5f;
    }
    atomicAdd(&clip[x * B_ + y], (ws / fmaxf(wsum, 1e-6f)) * (1.0f / NA_));
  }
  atomicAdd(&cnt, myCnt);
  atomicAdd(&srs, mySrs);
  __syncthreads();

  if (tid < B_) {
    float mx = -3.0e38f;
    for (int y = 0; y < B_; ++y) mx = fmaxf(mx, clip[tid * B_ + y]);
    float se = 0.0f;
    for (int y = 0; y < B_; ++y) se += expf(clip[tid * B_ + y] - mx);
    la[tid] = (mx + logf(se)) - clip[tid * B_ + tid];
  } else if (tid >= 32 && tid < 32 + B_) {
    int c = tid - 32;
    float mx = -3.0e38f;
    for (int x = 0; x < B_; ++x) mx = fmaxf(mx, clip[x * B_ + c]);
    float se = 0.0f;
    for (int x = 0; x < B_; ++x) se += expf(clip[x * B_ + c] - mx);
    lv[c] = (mx + logf(se)) - clip[c * B_ + c];
  }
  __syncthreads();

  if (tid == 0) {
    float s = 0.0f;
    for (int i = 0; i < B_; ++i) s += la[i] + lv[i];
    const float contrastive = s / (2.0f * B_);
    const float temp = temp_p[0];
    const double nsims = (double)B_ * B_ * NA_ * T_ * NV_;
    const float l_nonneg = (float)((double)nonneg_acc[0] / nsims);
    const float lt = logf(temp);
    float tl  = fmaxf(-lt, 0.0f);             tl  *= tl;
    float thi = fmaxf(lt - logf(4.0f), 0.0f); thi *= thi;
    float a1 = fmaxf(th - 0.9f, 0.0f), a2 = fmaxf(0.1f - th, 0.0f);
    float b1 = fmaxf(sf - 20.0f, 0.0f), b2 = fmaxf(1.0f - sf, 0.0f);
    const float reg = 0.15f * l_nonneg + 2.0f * (tl + thi)
                    + 0.1f * (a1 * a1 + a2 * a2) + 0.1f * (b1 * b1 + b2 * b2);
    const float frac   = cnt / (float)(B_ * B_ * NA_ * T_);
    const float selrew = -0.1f * log1pf(srs / (float)(B_ * NA_ * T_));
    out[0] = selrew + contrastive + reg;
    out[1] = contrastive;
    out[2] = reg;
    out[3] = frac;
    out[4] = selrew;
  }
}

// ---------------------------------------------------------------------------
extern "C" void kernel_launch(void* const* d_in, const int* in_sizes, int n_in,
                              void* d_out, int out_size, void* d_ws, size_t ws_size,
                              hipStream_t stream) {
  (void)in_sizes; (void)n_in; (void)out_size;
  const float* audio  = (const float*)d_in[0];
  const float* visual = (const float*)d_in[1];
  const float* temp   = (const float*)d_in[2];
  const float* scale  = (const float*)d_in[3];
  const float* thr    = (const float*)d_in[4];

  char*  wsb = (char*)d_ws;
  float* wsf = (float*)d_ws;
  dim3 grid((M_ + 63) / 64, GRP_);                  // (19, 240)

  init_kernel<<<1, 32, 0, stream>>>(wsf);

  if (ws_size >= A_END) {
    // Fast path: one-time normalize+split, then load/WMMA-only GEMM.
    __bf16* a_hi = (__bf16*)(wsb + A_AHI);
    __bf16* a_lo = (__bf16*)(wsb + A_ALO);
    __bf16* v_hi = (__bf16*)(wsb + A_VHI);
    __bf16* v_lo = (__bf16*)(wsb + A_VLO);
    float*  max_vis = (float*)(wsb + A_MAXV);

    normsplit_kernel<<<M_ + VR_, 32, 0, stream>>>(audio, visual,
                                                  a_hi, a_lo, v_hi, v_lo);
    simsmax_pre_kernel<<<grid, 128, 0, stream>>>(a_hi, a_lo, v_hi, v_lo, temp,
                                                 max_vis, wsf);
    finalize_kernel<<<1, 256, 0, stream>>>(max_vis, wsf, temp, scale, thr,
                                           (float*)d_out);
  } else {
    // Fallback: ~1.35 MB workspace, in-kernel conversion.
    float* a_rn    = wsf + WS_ARN;
    float* v_rn    = wsf + WS_VRN;
    float* max_vis = wsf + WS_MAXVIS;

    norm_kernel<<<M_ + VR_, 32, 0, stream>>>(audio, visual, a_rn, v_rn);
    simsmax_kernel<<<grid, 128, 0, stream>>>(audio, visual, a_rn, v_rn, temp,
                                             max_vis, wsf);
    finalize_kernel<<<1, 256, 0, stream>>>(max_vis, wsf, temp, scale, thr,
                                           (float*)d_out);
  }
}
